// AttentionBlockT_42923903156534
// MI455X (gfx1250) — matrix operand order repaired and checked by reading, not compiled, not run on hardware
//
#include <hip/hip_runtime.h>

// ---------------------------------------------------------------------------
// Types
// ---------------------------------------------------------------------------
typedef __attribute__((ext_vector_type(8)))  float          v8f;
typedef __attribute__((ext_vector_type(16))) __bf16         v16bf;
typedef __attribute__((ext_vector_type(4)))  unsigned int   v4u;
typedef __attribute__((ext_vector_type(4)))  float          v4f;

union BF16x16 {
    v16bf v;
    v4u   q[2];
    unsigned short s[16];
};

union PK16 {
    v4u q[2];
    unsigned short s[16];
};

// hardware f32 -> bf16 (let the backend pick v_cvt; avoids integer RNE emulation)
__device__ __forceinline__ unsigned short f2b(float f) {
    union { __bf16 b; unsigned short u; } cv;
    cv.b = (__bf16)f;
    return cv.u;
}

__device__ __forceinline__ v8f vzero8() {
    v8f z;
#pragma unroll
    for (int i = 0; i < 8; ++i) z[i] = 0.0f;
    return z;
}

__device__ __forceinline__ v8f wmma_bf16(const BF16x16& a, const BF16x16& b, v8f c) {
    return __builtin_amdgcn_wmma_f32_16x16x32_bf16(
        /*neg_a=*/false, a.v, /*neg_b=*/false, b.v,
        /*c_mod=*/(short)0, c, /*reuse_a=*/false, /*reuse_b=*/false);
}

#define BB   8
#define DD   512
#define TT   2048
#define KK   64

// ---------------------------------------------------------------------------
// Kernel 1: convert weights to bf16 (rows of 512), stash time-column fp32.
// ---------------------------------------------------------------------------
__global__ void prep_weights(const float* __restrict__ Wk, const float* __restrict__ Wq,
                             const float* __restrict__ Wv,
                             unsigned short* __restrict__ Wkb, unsigned short* __restrict__ Wqb,
                             unsigned short* __restrict__ Wvb,
                             float* __restrict__ wkt, float* __restrict__ wqt) {
    int i = blockIdx.x * blockDim.x + threadIdx.x;
    if (i < KK * DD) {
        int k = i >> 9, d = i & 511;
        Wkb[i] = f2b(Wk[k * (DD + 1) + d]);
        Wqb[i] = f2b(Wq[k * (DD + 1) + d]);
        Wvb[i] = f2b(Wv[k * DD + d]);
    }
    if (i < KK) {
        wkt[i] = Wk[i * (DD + 1) + DD];
        wqt[i] = Wq[i * (DD + 1) + DD];
    }
}

// ---------------------------------------------------------------------------
// Kernel 2: out[b, 0:512, :] = minibatch[b]  (the concat's x part)
// ---------------------------------------------------------------------------
__global__ void copy_x(const float* __restrict__ mb, float* __restrict__ out) {
    const int per_b = DD * TT / 4;                 // float4 per batch
    int i = blockIdx.x * blockDim.x + threadIdx.x;
    if (i < BB * per_b) {
        int b = i / per_b, r = i - b * per_b;
        const v4f* src = (const v4f*)mb;
        v4f* dst = (v4f*)(out + (size_t)b * (DD + KK) * TT);
        dst[r] = src[i];
    }
}

// ---------------------------------------------------------------------------
// Kernel 3: Q/K/V projections via WMMA. One wave per (b, 16-row t tile).
//   Q,K -> bf16 (B,T,64) row-major;  V -> bf16 (B,64,T) transposed via LDS.
// ---------------------------------------------------------------------------
__global__ __launch_bounds__(32) void proj_kernel(
    const float* __restrict__ mb,
    const unsigned short* __restrict__ Wqb, const unsigned short* __restrict__ Wkb,
    const unsigned short* __restrict__ Wvb,
    const float* __restrict__ wqt, const float* __restrict__ wkt,
    const float* __restrict__ bq, const float* __restrict__ bk, const float* __restrict__ bv,
    unsigned short* __restrict__ Qb, unsigned short* __restrict__ Kb,
    unsigned short* __restrict__ Vtb) {
    int wid  = blockIdx.x;
    int b    = wid >> 7;               // 128 t-tiles per batch
    int t0   = (wid & 127) << 4;
    int lane = threadIdx.x;
    int half = lane >> 4;
    int lan  = lane & 15;

    __shared__ __align__(16) float vlds[16 * 65];   // padded: bank-conflict-free columns

    v8f accQ[4], accK[4], accV[4];
#pragma unroll
    for (int n = 0; n < 4; ++n) { accQ[n] = vzero8(); accK[n] = vzero8(); accV[n] = vzero8(); }

    const float* xb = mb + (size_t)b * DD * TT;
    int trow = t0 + lan;

    for (int step = 0; step < 16; ++step) {          // d = step*32 .. +31
        // A-tile: x (16 t-rows x 32 d); lanes 0..15 cover consecutive t -> coalesced
        int dbase = step * 32 + half * 16;
        BF16x16 A;
#pragma unroll
        for (int i = 0; i < 8; ++i) {
            A.v[i]     = (__bf16)xb[(size_t)(dbase + i) * TT + trow];
            A.v[8 + i] = (__bf16)xb[(size_t)(dbase + 8 + i) * TT + trow];
        }
        int dcol = step * 32 + half * 16;

        // double-buffered B tiles: loads for n+1 overlap WMMAs for n
        BF16x16 Bq[2], Bk2[2], Bv2[2];
        {
            const v4u* pq = (const v4u*)(Wqb + lan * DD + dcol);
            Bq[0].q[0] = pq[0]; Bq[0].q[1] = pq[1];
            const v4u* pk = (const v4u*)(Wkb + lan * DD + dcol);
            Bk2[0].q[0] = pk[0]; Bk2[0].q[1] = pk[1];
            const v4u* pv = (const v4u*)(Wvb + lan * DD + dcol);
            Bv2[0].q[0] = pv[0]; Bv2[0].q[1] = pv[1];
        }
#pragma unroll
        for (int n = 0; n < 4; ++n) {
            int cur = n & 1, nxt = cur ^ 1;
            if (n < 3) {
                int krow = (n + 1) * 16 + lan;
                const v4u* pq = (const v4u*)(Wqb + krow * DD + dcol);
                Bq[nxt].q[0] = pq[0]; Bq[nxt].q[1] = pq[1];
                const v4u* pk = (const v4u*)(Wkb + krow * DD + dcol);
                Bk2[nxt].q[0] = pk[0]; Bk2[nxt].q[1] = pk[1];
                const v4u* pv = (const v4u*)(Wvb + krow * DD + dcol);
                Bv2[nxt].q[0] = pv[0]; Bv2[nxt].q[1] = pv[1];
            }
            accQ[n] = wmma_bf16(A, Bq[cur],  accQ[n]);
            accK[n] = wmma_bf16(A, Bk2[cur], accK[n]);
            accV[n] = wmma_bf16(A, Bv2[cur], accV[n]);
        }
    }

    // epilogue: Q/K get time-feature + bias, stored row-major (coalesced across lanes)
#pragma unroll
    for (int n = 0; n < 4; ++n) {
#pragma unroll
        for (int r = 0; r < 8; ++r) {
            int M = r + half * 8;
            int t = t0 + M;
            int k = n * 16 + lan;
            float tf = (float)(t + 1);
            Qb[((size_t)b * TT + t) * KK + k] = f2b(accQ[n][r] + tf * wqt[k] + bq[k]);
            Kb[((size_t)b * TT + t) * KK + k] = f2b(accK[n][r] + tf * wkt[k] + bk[k]);
            vlds[M * 65 + k] = accV[n][r] + bv[k];
        }
    }

    // V transpose: each lane packs 2 k-columns x 16 t into b128 stores
#pragma unroll
    for (int c = 0; c < 2; ++c) {
        int k = lane + c * 32;
        PK16 pack;
#pragma unroll
        for (int t = 0; t < 16; ++t) pack.s[t] = f2b(vlds[t * 65 + k]);
        v4u* dst = (v4u*)(Vtb + ((size_t)b * KK + k) * TT + t0);
        dst[0] = pack.q[0];
        dst[1] = pack.q[1];
    }
}

// ---------------------------------------------------------------------------
// Kernel 4: causal flash attention. One wave per (b, 16-query tile),
// 32 keys per inner iteration, online softmax in exp2 domain.
// ---------------------------------------------------------------------------
__global__ __launch_bounds__(32) void attn_kernel(
    const unsigned short* __restrict__ Qb, const unsigned short* __restrict__ Kb,
    const unsigned short* __restrict__ Vtb, float* __restrict__ out) {
    const float kscale = 0.125f * 1.44269504088896f;   // (1/sqrt(64)) * log2(e)
    int wid  = blockIdx.x;
    int b    = wid >> 7;
    int t0   = (wid & 127) << 4;
    int lane = threadIdx.x;
    int half = lane >> 4;
    int lan  = lane & 15;

    __shared__ __align__(16) unsigned short Plds[16 * 32];

    const unsigned short* kbase = Kb + (size_t)b * TT * KK;
    const unsigned short* vbase = Vtb + (size_t)b * KK * TT;

    // Q tile (16x64) as two A-layout bf16 fragments (K-steps of 32)
    BF16x16 Aq[2];
    {
        const unsigned short* qrow = Qb + ((size_t)b * TT + t0 + lan) * KK;
#pragma unroll
        for (int st = 0; st < 2; ++st) {
            Aq[st].q[0] = *(const v4u*)(qrow + st * 32 + half * 16);
            Aq[st].q[1] = *(const v4u*)(qrow + st * 32 + half * 16 + 8);
        }
    }

    v8f O[4];
#pragma unroll
    for (int n = 0; n < 4; ++n) O[n] = vzero8();
    float m[8], l[8];
#pragma unroll
    for (int r = 0; r < 8; ++r) { m[r] = -1e30f; l[r] = 0.0f; }

    for (int s0 = 0; s0 < t0 + 16; s0 += 32) {
        // prefetch next key/value block (global_prefetch_b8)
        if (s0 + 32 < t0 + 16) {
            __builtin_prefetch(kbase + (size_t)(s0 + 32 + lane) * KK, 0, 1);
            __builtin_prefetch(vbase + (size_t)lane * TT + s0 + 32, 0, 1);
        }

        // ---- load all 4 key fragments, then 4 WMMAs (one wait, overlapped) ----
        BF16x16 BkA[2], BkB[2];                       // [j] = 16-key half; A/B = k-step
#pragma unroll
        for (int j = 0; j < 2; ++j) {
            const v4u* kr = (const v4u*)(kbase + (size_t)(s0 + j * 16 + lan) * KK + half * 16);
            BkA[j].q[0] = kr[0]; BkA[j].q[1] = kr[1]; // features 0..31 slice
            BkB[j].q[0] = kr[4]; BkB[j].q[1] = kr[5]; // features 32..63 slice
        }
        v8f S0 = vzero8(), S1 = vzero8();
        S0 = wmma_bf16(Aq[0], BkA[0], S0);
        S0 = wmma_bf16(Aq[1], BkB[0], S0);
        S1 = wmma_bf16(Aq[0], BkA[1], S1);
        S1 = wmma_bf16(Aq[1], BkB[1], S1);

        bool needMask = (s0 + 31 > t0);
        // ---- online softmax update, write P (bf16) to LDS in row-major ----
#pragma unroll
        for (int r = 0; r < 8; ++r) {
            int M = r + half * 8;
            float s0v = S0[r] * kscale;
            float s1v = S1[r] * kscale;
            if (needMask) {
                int t = t0 + M;
                if (s0 + lan > t)      s0v = -1e30f;
                if (s0 + 16 + lan > t) s1v = -1e30f;
            }
            float mx = fmaxf(s0v, s1v);
            mx = fmaxf(mx, __shfl_xor(mx, 1));
            mx = fmaxf(mx, __shfl_xor(mx, 2));
            mx = fmaxf(mx, __shfl_xor(mx, 4));
            mx = fmaxf(mx, __shfl_xor(mx, 8));
            float mnew  = fmaxf(m[r], mx);
            float alpha = exp2f(m[r] - mnew);
            float p0 = exp2f(s0v - mnew);
            float p1 = exp2f(s1v - mnew);
            float rs = p0 + p1;
            rs += __shfl_xor(rs, 1);
            rs += __shfl_xor(rs, 2);
            rs += __shfl_xor(rs, 4);
            rs += __shfl_xor(rs, 8);
            l[r] = l[r] * alpha + rs;
            m[r] = mnew;
#pragma unroll
            for (int n = 0; n < 4; ++n) O[n][r] *= alpha;
            Plds[M * 32 + lan]      = f2b(p0);
            Plds[M * 32 + 16 + lan] = f2b(p1);
        }

        // ---- read P back in A-layout (C->A conversion via LDS) ----
        BF16x16 Pa;
        Pa.q[0] = *(const v4u*)(Plds + lan * 32 + half * 16);
        Pa.q[1] = *(const v4u*)(Plds + lan * 32 + half * 16 + 8);

        // ---- load all 4 V fragments, then 4 WMMAs ----
        BF16x16 Bv[4];
#pragma unroll
        for (int n = 0; n < 4; ++n) {
            const v4u* vr = (const v4u*)(vbase + (size_t)(n * 16 + lan) * TT + s0 + half * 16);
            Bv[n].q[0] = vr[0]; Bv[n].q[1] = vr[1];
        }
#pragma unroll
        for (int n = 0; n < 4; ++n) O[n] = wmma_bf16(Pa, Bv[n], O[n]);
    }

    // ---- normalize and write out rows 512..575 of (B, 576, T), b128 stores ----
    float inv[8];
#pragma unroll
    for (int r = 0; r < 8; ++r) inv[r] = 1.0f / l[r];
#pragma unroll
    for (int n = 0; n < 4; ++n) {
        int v = n * 16 + lan;
        float* dst = out + ((size_t)b * (DD + KK) + DD + v) * TT + t0 + half * 8;
        v4f lo = { O[n][0] * inv[0], O[n][1] * inv[1], O[n][2] * inv[2], O[n][3] * inv[3] };
        v4f hi = { O[n][4] * inv[4], O[n][5] * inv[5], O[n][6] * inv[6], O[n][7] * inv[7] };
        ((v4f*)dst)[0] = lo;
        ((v4f*)dst)[1] = hi;
    }
}

// ---------------------------------------------------------------------------
// Launch
// ---------------------------------------------------------------------------
extern "C" void kernel_launch(void* const* d_in, const int* in_sizes, int n_in,
                              void* d_out, int out_size, void* d_ws, size_t ws_size,
                              hipStream_t stream) {
    const float* mb = (const float*)d_in[0];
    const float* Wk = (const float*)d_in[1];
    const float* bk = (const float*)d_in[2];
    const float* Wq = (const float*)d_in[3];
    const float* bq = (const float*)d_in[4];
    const float* Wv = (const float*)d_in[5];
    const float* bv = (const float*)d_in[6];
    float* out = (float*)d_out;

    char* ws = (char*)d_ws;
    unsigned short* Qb  = (unsigned short*)(ws);                           // 2 MB
    unsigned short* Kb  = (unsigned short*)(ws + (size_t)2 * 1024 * 1024); // 2 MB
    unsigned short* Vtb = (unsigned short*)(ws + (size_t)4 * 1024 * 1024); // 2 MB
    unsigned short* Wqb = (unsigned short*)(ws + (size_t)6 * 1024 * 1024); // 64 KB
    unsigned short* Wkb = Wqb + 64 * 512;                                  // 64 KB
    unsigned short* Wvb = Wkb + 64 * 512;                                  // 64 KB
    float* wqt = (float*)(Wvb + 64 * 512);
    float* wkt = wqt + 64;

    prep_weights<<<(64 * 512 + 255) / 256, 256, 0, stream>>>(Wk, Wq, Wv, Wkb, Wqb, Wvb, wkt, wqt);
    copy_x<<<(8 * 512 * 2048 / 4 + 255) / 256, 256, 0, stream>>>(mb, out);
    proj_kernel<<<8 * 128, 32, 0, stream>>>(mb, Wqb, Wkb, Wvb, wqt, wkt, bq, bk, bv, Qb, Kb, Vtb);
    attn_kernel<<<8 * 128, 32, 0, stream>>>(Qb, Kb, Vtb, out);
}
